// E_GCL_78065325572140
// MI455X (gfx1250) — matrix operand order repaired and checked by reading, not compiled, not run on hardware
//
#include <hip/hip_runtime.h>
#include <hip/hip_bf16.h>

typedef __attribute__((ext_vector_type(8)))  _Float16 v8h;
typedef __attribute__((ext_vector_type(16))) _Float16 v16h;
typedef __attribute__((ext_vector_type(8)))  float    v8f;

#define N_LAT 121
#define N_LON 240
#define NNODE (N_LAT * N_LON)      // 29040
#define DEG   16
#define NEDGE (NNODE * DEG)        // 464640
#define HD    128
#define LVN   11
#define EK    320                  // edge-in K padded 290 -> 320

// packed weight sizes (halves): Kt * Nt * 512
#define SZ_WE1 (10 * 8 * 512)
#define SZ_WE2 (4  * 8 * 512)
#define SZ_WN1 (12 * 8 * 512)
#define SZ_WN2 (4  * 8 * 512)
#define SZ_WC1 (4  * 8 * 512)
#define SZ_WCL (4  * 2 * 512)

#define EW  4                      // waves/block, edge kernel
#define NWV 2                      // waves/block, node kernel
#define NTILES ((NNODE + 15) / 16) // 1815

static_assert(NEDGE % (16 * EW) == 0, "edge grid must be exact");

// ---------------- weight pack: fp32 row-major [K,N] -> f16 WMMA B fragments ----------------
// fragment (kt,nt): 32 lanes x 16 halves; lane holds column n = nt*16 + lane%16,
// half i holds k = kt*32 + (lane>=16)*16 + i   (contiguous k order, same as mm_tile's A loads)
__global__ void pack_w(const float* __restrict__ W, _Float16* __restrict__ dst,
                       int Kreal, int Nreal, int Kt, int Nt) {
  int idx = blockIdx.x * blockDim.x + threadIdx.x;
  int total = Kt * Nt * 512;
  if (idx >= total) return;
  int i    = idx & 15;
  int lane = (idx >> 4) & 31;
  int fr   = idx >> 9;             // kt*Nt + nt
  int nt   = fr % Nt;
  int kt   = fr / Nt;
  int n = nt * 16 + (lane & 15);
  int k = kt * 32 + ((lane >> 4) << 4) + i;
  float val = (k < Kreal && n < Nreal) ? W[k * Nreal + n] : 0.0f;
  dst[idx] = (_Float16)val;
}

// ---------------- WMMA M=16 tile GEMM, software-pipelined (double-buffered fragments) ----
// A: row-major f16 (LDS), stride in halves. B: packed fragments (global, L2-resident).
// Loads for step kt+1 are issued before the WMMA of step kt so fragment-load latency
// overlaps WMMA execution instead of stalling on s_wait before every WMMA.
__device__ __forceinline__ v8f mm_tile(const _Float16* __restrict__ A, int aStride,
                                       const _Float16* __restrict__ B,
                                       int ktCount, int nt, int Nt, v8f acc, int lane) {
  const int base = (lane >> 4) << 4;               // A fragment: 0 or 16
  const _Float16* aRow = A + (lane & 15) * aStride;

  v16h a, b;
  *(v8h*)&a       = *(const v8h*)(aRow + base);
  *((v8h*)&a + 1) = *(const v8h*)(aRow + 8 + base);
  b = *(const v16h*)(B + ((size_t)nt * 32 + lane) * 16);

#pragma unroll
  for (int kt = 0; kt < ktCount; ++kt) {
    v16h an = a, bn = b;
    if (kt + 1 < ktCount) {                        // prefetch next fragments
      *(v8h*)&an       = *(const v8h*)(aRow + (kt + 1) * 32 + base);
      *((v8h*)&an + 1) = *(const v8h*)(aRow + (kt + 1) * 32 + 8 + base);
      bn = *(const v16h*)(B + ((size_t)((kt + 1) * Nt + nt) * 32 + lane) * 16);
    }
    acc = __builtin_amdgcn_wmma_f32_16x16x32_f16(false, a, false, b, (short)0, acc,
                                                 false, false);
    a = an;
    b = bn;
  }
  return acc;
}

// ---------------- fused edge pipeline: 1 wave = 16 edges ----------------
__global__ __launch_bounds__(EW * 32) void edge_kernel(
    const float* __restrict__ h, const float* __restrict__ u, const float* __restrict__ v,
    const float* __restrict__ eattr, const int* __restrict__ eidx,
    const _Float16* __restrict__ pWe1, const float* __restrict__ be1,
    const _Float16* __restrict__ pWe2, const float* __restrict__ be2,
    const _Float16* __restrict__ pWc1, const float* __restrict__ bc1,
    const _Float16* __restrict__ pWcl,
    float* __restrict__ agg, float* __restrict__ sum_u, float* __restrict__ sum_v,
    float* __restrict__ cnt)
{
  __shared__ alignas(16) _Float16 s_in[EW][16 * EK];   // edge_in; reused for edge_feat
  __shared__ alignas(16) _Float16 s_hid[EW][16 * HD];  // hidden; reused for coord t
  __shared__ float s_rad2[EW][16][2];
  __shared__ int   s_row[EW][16];
  __shared__ int   s_col[EW][16];

  const int lane = threadIdx.x & 31;
  const int w    = threadIdx.x >> 5;
  const int tile = blockIdx.x * EW + w;   // grid is exact
  const int e0   = tile * 16;
  _Float16* sin_ = s_in[w];
  _Float16* shid = s_hid[w];

  if (lane < 16) {
    s_row[w][lane] = eidx[e0 + lane];
    s_col[w][lane] = eidx[NEDGE + e0 + lane];
  }
  __syncthreads();

  // ---- stage edge_in = [h_row | h_col | rel_dirt | col_speed | row_speed | attr | 0-pad] ----
  for (int j = lane; j < 16 * HD; j += 32) {
    int m = j >> 7, d = j & (HD - 1);
    sin_[m * EK + d]      = (_Float16)h[s_row[w][m] * HD + d];
    sin_[m * EK + HD + d] = (_Float16)h[s_col[w][m] * HD + d];
  }
  for (int j = lane; j < 16 * LVN; j += 32) {
    int m = j / LVN, l = j - m * LVN;
    int ci = s_col[w][m], ri = s_row[w][m];
    float uc = u[ci * LVN + l], vc = v[ci * LVN + l];
    float ur = u[ri * LVN + l], vr = v[ri * LVN + l];
    float cs = sqrtf(uc * uc + vc * vc);
    float rs = sqrtf(ur * ur + vr * vr);
    float rd = (uc * ur + vc * vr) / (cs * rs);
    sin_[m * EK + 256 + l] = (_Float16)rd;
    sin_[m * EK + 267 + l] = (_Float16)cs;
    sin_[m * EK + 278 + l] = (_Float16)rs;
  }
  if (lane < 16) sin_[lane * EK + 289] = (_Float16)eattr[e0 + lane];
  for (int j = lane; j < 16 * 30; j += 32) {
    int m = j / 30, p = 290 + (j - m * 30);
    sin_[m * EK + p] = (_Float16)0.0f;
  }
  { // per-edge rad2 (uses col wind only)
    int m = lane & 15, comp = lane >> 4;
    const float* src = comp ? v : u;
    int ci = s_col[w][m];
    float s = 0.f;
    for (int l = 0; l < LVN; ++l) { float x = src[ci * LVN + l]; s += x * x; }
    s_rad2[w][m][comp] = sqrtf(s);
  }
  if (lane < 16) atomicAdd(&cnt[s_row[w][lane]], 1.0f);
  __syncthreads();

  const int mB = (lane >> 4) << 3;   // C/D: row base 0 or 8
  const int nL = lane & 15;          // C/D: column within n-tile

  // ---- layer 1: edge_in[16x320] @ W_e1 -> relu -> hidden ----
  for (int nt = 0; nt < 8; ++nt) {
    float b = be1[nt * 16 + nL];
    v8f acc = {b, b, b, b, b, b, b, b};
    acc = mm_tile(sin_, EK, pWe1, 10, nt, 8, acc, lane);
    int n = nt * 16 + nL;
    for (int r = 0; r < 8; ++r)
      shid[(mB + r) * HD + n] = (_Float16)fmaxf(acc[r], 0.f);
  }
  __syncthreads();

  // ---- layer 2: hidden @ W_e2 -> relu -> edge_feat (into s_in) + scatter-sum agg ----
  for (int nt = 0; nt < 8; ++nt) {
    float b = be2[nt * 16 + nL];
    v8f acc = {b, b, b, b, b, b, b, b};
    acc = mm_tile(shid, HD, pWe2, 4, nt, 8, acc, lane);
    int n = nt * 16 + nL;
    for (int r = 0; r < 8; ++r) {
      float x = fmaxf(acc[r], 0.f);
      sin_[(mB + r) * EK + n] = (_Float16)x;
      atomicAdd(&agg[s_row[w][mB + r] * HD + n], x);
    }
  }
  __syncthreads();

  // ---- layer 3: edge_feat @ W_c1 -> relu -> t (into s_hid) ----
  for (int nt = 0; nt < 8; ++nt) {
    float b = bc1[nt * 16 + nL];
    v8f acc = {b, b, b, b, b, b, b, b};
    acc = mm_tile(sin_, EK, pWc1, 4, nt, 8, acc, lane);
    int n = nt * 16 + nL;
    for (int r = 0; r < 8; ++r)
      shid[(mB + r) * HD + n] = (_Float16)fmaxf(acc[r], 0.f);
  }
  __syncthreads();

  // ---- layer 4: t @ W_cl[128x22] -> ef; wind = ef * rad2; scatter-sum ----
  for (int nt = 0; nt < 2; ++nt) {
    v8f acc = {0.f, 0.f, 0.f, 0.f, 0.f, 0.f, 0.f, 0.f};
    acc = mm_tile(shid, HD, pWcl, 4, nt, 2, acc, lane);
    int n = nt * 16 + nL;
    if (n < 2 * LVN) {
      int comp = n / LVN, l = n - comp * LVN;
      float* dst = comp ? sum_v : sum_u;
      for (int r = 0; r < 8; ++r) {
        int m = mB + r;
        float wv = acc[r] * s_rad2[w][m][comp];
        atomicAdd(&dst[s_row[w][m] * LVN + l], wv);
      }
    }
  }
}

// ---------------- latitude-band mean of agg ----------------
__global__ void lat_mean_kernel(const float* __restrict__ agg, float* __restrict__ latm) {
  int idx = blockIdx.x * blockDim.x + threadIdx.x;
  if (idx >= N_LAT * HD) return;
  int lat = idx >> 7, d = idx & 127;
  float s = 0.f;
  const float* p = agg + (size_t)lat * N_LON * HD + d;
  for (int lon = 0; lon < N_LON; ++lon) s += p[lon * HD];
  latm[idx] = s * (1.0f / N_LON);
}

// ---------------- node MLP + residual: 1 wave = 16 nodes ----------------
__global__ __launch_bounds__(NWV * 32) void node_kernel(
    const float* __restrict__ h, const float* __restrict__ agg,
    const float* __restrict__ latm,
    const _Float16* __restrict__ pWn1, const float* __restrict__ bn1,
    const _Float16* __restrict__ pWn2, const float* __restrict__ bn2,
    float* __restrict__ out)
{
  __shared__ alignas(16) _Float16 s_in[NWV][16 * 384];
  __shared__ alignas(16) _Float16 s_hid[NWV][16 * HD];
  const int lane  = threadIdx.x & 31;
  const int w     = threadIdx.x >> 5;
  const int tileR = blockIdx.x * NWV + w;
  const bool valid = tileR < NTILES;
  const int tile = valid ? tileR : NTILES - 1;   // keep __syncthreads uniform
  _Float16* sin_ = s_in[w];
  _Float16* shid = s_hid[w];
  const int n0 = tile * 16;

  for (int j = lane; j < 16 * HD; j += 32) {
    int m = j >> 7, d = j & (HD - 1);
    int node = n0 + m;
    sin_[m * 384 + d]       = (_Float16)h[node * HD + d];
    sin_[m * 384 + 128 + d] = (_Float16)agg[node * HD + d];
    sin_[m * 384 + 256 + d] = (_Float16)latm[(node / N_LON) * HD + d];
  }
  __syncthreads();

  const int mB = (lane >> 4) << 3;
  const int nL = lane & 15;

  for (int nt = 0; nt < 8; ++nt) {
    float b = bn1[nt * 16 + nL];
    v8f acc = {b, b, b, b, b, b, b, b};
    acc = mm_tile(sin_, 384, pWn1, 12, nt, 8, acc, lane);
    int n = nt * 16 + nL;
    for (int r = 0; r < 8; ++r)
      shid[(mB + r) * HD + n] = (_Float16)fmaxf(acc[r], 0.f);
  }
  __syncthreads();

  for (int nt = 0; nt < 8; ++nt) {
    float b = bn2[nt * 16 + nL];
    v8f acc = {b, b, b, b, b, b, b, b};
    acc = mm_tile(shid, HD, pWn2, 4, nt, 8, acc, lane);
    int n = nt * 16 + nL;
    if (valid) {
      for (int r = 0; r < 8; ++r) {
        int node = n0 + mB + r;
        out[node * HD + n] = acc[r] + h[node * HD + n];   // residual
      }
    }
  }
}

// ---------------- finalize coord outputs: mean + clip ----------------
__global__ void coord_out_kernel(const float* __restrict__ su, const float* __restrict__ sv,
                                 const float* __restrict__ cnt, float* __restrict__ out) {
  int idx = blockIdx.x * blockDim.x + threadIdx.x;
  if (idx >= NNODE * LVN) return;
  int node = idx / LVN;
  float c = fmaxf(cnt[node], 1.0f);
  float a = fminf(fmaxf(su[idx] / c, -100.f), 100.f);
  float b = fminf(fmaxf(sv[idx] / c, -100.f), 100.f);
  out[(size_t)NNODE * HD + idx] = a;
  out[(size_t)NNODE * HD + (size_t)NNODE * LVN + idx] = b;
}

__global__ void zero_kernel(float* __restrict__ p, int n) {
  int i = blockIdx.x * blockDim.x + threadIdx.x;
  if (i < n) p[i] = 0.f;
}

extern "C" void kernel_launch(void* const* d_in, const int* in_sizes, int n_in,
                              void* d_out, int out_size, void* d_ws, size_t ws_size,
                              hipStream_t stream) {
  const float* h   = (const float*)d_in[0];
  const float* u   = (const float*)d_in[1];
  const float* v   = (const float*)d_in[2];
  const float* ea  = (const float*)d_in[3];
  const float* We1 = (const float*)d_in[4];
  const float* be1 = (const float*)d_in[5];
  const float* We2 = (const float*)d_in[6];
  const float* be2 = (const float*)d_in[7];
  const float* Wn1 = (const float*)d_in[8];
  const float* bn1 = (const float*)d_in[9];
  const float* Wn2 = (const float*)d_in[10];
  const float* bn2 = (const float*)d_in[11];
  const float* Wc1 = (const float*)d_in[12];
  const float* bc1 = (const float*)d_in[13];
  const float* Wcl = (const float*)d_in[14];
  const int*  eidx = (const int*)d_in[15];
  float* out = (float*)d_out;

  // workspace carve-up
  char* ws = (char*)d_ws;
  size_t off = 0;
  auto takeH = [&](size_t halves) { _Float16* p = (_Float16*)(ws + off); off += halves * 2; return p; };
  _Float16* pWe1 = takeH(SZ_WE1);
  _Float16* pWe2 = takeH(SZ_WE2);
  _Float16* pWn1 = takeH(SZ_WN1);
  _Float16* pWn2 = takeH(SZ_WN2);
  _Float16* pWc1 = takeH(SZ_WC1);
  _Float16* pWcl = takeH(SZ_WCL);
  auto takeF = [&](size_t n) { float* p = (float*)(ws + off); off += n * 4; return p; };
  float* agg  = takeF((size_t)NNODE * HD);
  float* su   = takeF((size_t)NNODE * LVN);
  float* sv   = takeF((size_t)NNODE * LVN);
  float* cnt  = takeF(NNODE);
  float* latm = takeF((size_t)N_LAT * HD);

  auto pack = [&](const float* W, _Float16* dst, int Kr, int Nr, int Kt, int Nt) {
    int total = Kt * Nt * 512;
    pack_w<<<(total + 255) / 256, 256, 0, stream>>>(W, dst, Kr, Nr, Kt, Nt);
  };
  pack(We1, pWe1, 2 * HD + 34, HD, 10, 8);
  pack(We2, pWe2, HD, HD, 4, 8);
  pack(Wn1, pWn1, 3 * HD, HD, 12, 8);
  pack(Wn2, pWn2, HD, HD, 4, 8);
  pack(Wc1, pWc1, HD, HD, 4, 8);
  pack(Wcl, pWcl, HD, 2 * LVN, 4, 2);

  // zero agg | su | sv | cnt (contiguous)
  int nz = NNODE * HD + 2 * NNODE * LVN + NNODE;
  zero_kernel<<<(nz + 255) / 256, 256, 0, stream>>>(agg, nz);

  edge_kernel<<<NEDGE / 16 / EW, EW * 32, 0, stream>>>(
      h, u, v, ea, eidx, pWe1, be1, pWe2, be2, pWc1, bc1, pWcl,
      agg, su, sv, cnt);

  lat_mean_kernel<<<(N_LAT * HD + 255) / 256, 256, 0, stream>>>(agg, latm);

  node_kernel<<<(NTILES + NWV - 1) / NWV, NWV * 32, 0, stream>>>(
      h, agg, latm, pWn1, bn1, pWn2, bn2, out);

  coord_out_kernel<<<(NNODE * LVN + 255) / 256, 256, 0, stream>>>(su, sv, cnt, out);
}
